// AttentivePoolingLayer_15204184228068
// MI455X (gfx1250) — matrix operand order repaired and checked by reading, not compiled, not run on hardware
//
#include <hip/hip_runtime.h>
#include <hip/hip_bf16.h>
#include <math.h>

#define Bc 16
#define Tc 2048
#define Dc 256

typedef __attribute__((ext_vector_type(16))) _Float16 v16h;
typedef __attribute__((ext_vector_type(8)))  _Float16 v8h;
typedef __attribute__((ext_vector_type(8)))  float    v8f;

// ---------------------------------------------------------------------------
// helpers
// ---------------------------------------------------------------------------

// sign-split float atomic max (exact, order-independent). Requires init = -inf.
__device__ __forceinline__ void atomicMaxF(float* addr, float val) {
    if (val >= 0.0f) atomicMax((int*)addr, __float_as_int(val));
    else             atomicMin((unsigned int*)addr, __float_as_uint(val));
}

// Load 16 f32 (two contiguous 8-float segments) and convert to a v16h fragment.
__device__ __forceinline__ v16h load16_f32_to_h(const float* __restrict__ p,
                                                int off0, int off1) {
    const float4* s0 = (const float4*)(p + off0);
    const float4* s1 = (const float4*)(p + off1);
    float4 f0 = s0[0], f1 = s0[1], f2 = s1[0], f3 = s1[1];
    v16h r;
    r[0]  = (_Float16)f0.x; r[1]  = (_Float16)f0.y; r[2]  = (_Float16)f0.z; r[3]  = (_Float16)f0.w;
    r[4]  = (_Float16)f1.x; r[5]  = (_Float16)f1.y; r[6]  = (_Float16)f1.z; r[7]  = (_Float16)f1.w;
    r[8]  = (_Float16)f2.x; r[9]  = (_Float16)f2.y; r[10] = (_Float16)f2.z; r[11] = (_Float16)f2.w;
    r[12] = (_Float16)f3.x; r[13] = (_Float16)f3.y; r[14] = (_Float16)f3.z; r[15] = (_Float16)f3.w;
    return r;
}

// ---------------------------------------------------------------------------
// 0a) U (f32, row-major [k][n]) -> Uh (f16, column-major: Uh[n*D + k])
// ---------------------------------------------------------------------------
__global__ void k_prepU(const float* __restrict__ U, _Float16* __restrict__ Uh) {
    int idx = blockIdx.x * blockDim.x + threadIdx.x;
    if (idx >= Dc * Dc) return;
    int n = idx >> 8;       // column of U
    int k = idx & 255;      // row of U
    Uh[idx] = (_Float16)U[k * Dc + n];
}

// 0b) a (f32) -> ah (f16), streaming, 8 elements/thread
__global__ void __launch_bounds__(256)
k_convA(const float* __restrict__ a, _Float16* __restrict__ ah) {
    size_t i = ((size_t)blockIdx.x * blockDim.x + threadIdx.x) * 8;
    const float4* s = (const float4*)(a + i);
    float4 f0 = s[0], f1 = s[1];
    v8h r;
    r[0] = (_Float16)f0.x; r[1] = (_Float16)f0.y; r[2] = (_Float16)f0.z; r[3] = (_Float16)f0.w;
    r[4] = (_Float16)f1.x; r[5] = (_Float16)f1.y; r[6] = (_Float16)f1.z; r[7] = (_Float16)f1.w;
    *(v8h*)(ah + i) = r;
}

__global__ void k_initNegInf(float* __restrict__ p, int n) {
    int i = blockIdx.x * blockDim.x + threadIdx.x;
    if (i < n) p[i] = -__builtin_inff();
}

// ---------------------------------------------------------------------------
// 1) qUh[b,t,:] = f16( q[b,t,:] @ U )   via v_wmma_f32_16x16x32_f16
//    grid: (T/16, D/64, B), block: 128 (4 waves; wave w -> N-tile y*4+w)
// ---------------------------------------------------------------------------
__global__ void __launch_bounds__(128)
k_qU(const float* __restrict__ q, const _Float16* __restrict__ Uh,
     _Float16* __restrict__ qUh) {
    const int lane  = threadIdx.x & 31;
    const int wave  = threadIdx.x >> 5;
    const int m     = lane & 15;         // A row / C column
    const int hi    = lane >> 4;         // half-wave select
    const int b     = blockIdx.z;
    const int Mbase = blockIdx.x * 16;
    const int Ncol  = (blockIdx.y * 4 + wave) * 16 + m;

    const float*    qrow = q  + (size_t)(b * Tc + Mbase + m) * Dc;
    const _Float16* ucol = Uh + (size_t)Ncol * Dc;   // column-major

    v8f c = {};
#pragma unroll
    for (int kk = 0; kk < Dc; kk += 32) {
        // A fragment: K = kk + hi*16 + 0..15 (contiguous, matches B)
        v16h av = load16_f32_to_h(qrow, kk + hi * 16, kk + hi * 16 + 8);
        // B fragment: column Ncol, K = kk + hi*16 + 0..15 (contiguous f16)
        v16h bv = *(const v16h*)(ucol + kk + hi * 16);
        c = __builtin_amdgcn_wmma_f32_16x16x32_f16(false, av, false, bv,
                                                   (short)0, c, false, false);
    }
    // C layout: lane holds rows r + 8*hi, column m
#pragma unroll
    for (int r = 0; r < 8; ++r) {
        int row = Mbase + r + 8 * hi;
        qUh[(size_t)(b * Tc + row) * Dc + Ncol] = (_Float16)c[r];
    }
}

// ---------------------------------------------------------------------------
// 2) Fused pre-activation maxes of G = qUh @ ah^T.
//    tanh is monotonic => max(tanh(x)) = tanh(max(x)); tanh applied later.
//    grid: (T/16, B), block: 256 = 8 waves; each wave owns 16 N-tiles,
//    processed as 8 iterations of TWO independent accumulator chains (ILP).
// ---------------------------------------------------------------------------
__global__ void __launch_bounds__(256)
k_G(const _Float16* __restrict__ ah, const _Float16* __restrict__ qUh,
    float* __restrict__ rowmaxG, float* __restrict__ colmaxG) {
    __shared__ float smax[8 * 16];
    const int lane  = threadIdx.x & 31;
    const int wave  = threadIdx.x >> 5;
    const int m     = lane & 15;
    const int hi    = lane >> 4;
    const int b     = blockIdx.y;
    const int Mbase = blockIdx.x * 16;

    // Preload all A fragments of this M-tile (K = 0..255): 8 x v16h = 64 VGPRs
    v16h afr[8];
    const _Float16* qrow = qUh + (size_t)(b * Tc + Mbase + m) * Dc;
#pragma unroll
    for (int kki = 0; kki < 8; ++kki) {
        const int kk = kki * 32;
        v8h lo = *(const v8h*)(qrow + kk + hi * 16);
        v8h up = *(const v8h*)(qrow + kk + hi * 16 + 8);
#pragma unroll
        for (int e = 0; e < 8; ++e) { afr[kki][e] = lo[e]; afr[kki][8 + e] = up[e]; }
    }

    v8f rmax;
#pragma unroll
    for (int r = 0; r < 8; ++r) rmax[r] = -__builtin_inff();

    for (int nt = wave * 2; nt < Tc / 16; nt += 16) {
        const int cols0 = nt * 16;
        const int cols1 = cols0 + 16;
        const _Float16* arow0 = ah + (size_t)(b * Tc + cols0 + m) * Dc + hi * 16;
        const _Float16* arow1 = ah + (size_t)(b * Tc + cols1 + m) * Dc + hi * 16;
        v8f c0 = {}, c1 = {};
#pragma unroll
        for (int kki = 0; kki < 8; ++kki) {
            const int kk = kki * 32;
            v16h b0 = *(const v16h*)(arow0 + kk);
            v16h b1 = *(const v16h*)(arow1 + kk);
            c0 = __builtin_amdgcn_wmma_f32_16x16x32_f16(false, afr[kki], false, b0,
                                                        (short)0, c0, false, false);
            c1 = __builtin_amdgcn_wmma_f32_16x16x32_f16(false, afr[kki], false, b1,
                                                        (short)0, c1, false, false);
        }
        float cm0 = -__builtin_inff(), cm1 = -__builtin_inff();
#pragma unroll
        for (int r = 0; r < 8; ++r) {
            rmax[r] = fmaxf(rmax[r], fmaxf(c0[r], c1[r]));
            cm0     = fmaxf(cm0, c0[r]);
            cm1     = fmaxf(cm1, c1[r]);
        }
        // column max over 16 rows: merge the two half-wave copies of column m
        cm0 = fmaxf(cm0, __shfl_xor(cm0, 16, 32));
        cm1 = fmaxf(cm1, __shfl_xor(cm1, 16, 32));
        if (lane < 16) {
            atomicMaxF(&colmaxG[b * Tc + cols0 + lane], cm0);
            atomicMaxF(&colmaxG[b * Tc + cols1 + lane], cm1);
        }
    }

    // row max: reduce across the 16 lanes of each half-wave
#pragma unroll
    for (int mask = 8; mask >= 1; mask >>= 1)
#pragma unroll
        for (int r = 0; r < 8; ++r)
            rmax[r] = fmaxf(rmax[r], __shfl_xor(rmax[r], mask, 32));
    if (m == 0) {                          // lanes 0 (rows 0-7) and 16 (rows 8-15)
#pragma unroll
        for (int r = 0; r < 8; ++r) smax[wave * 16 + hi * 8 + r] = rmax[r];
    }
    __syncthreads();
    if (threadIdx.x < 16) {
        float v = smax[threadIdx.x];
#pragma unroll
        for (int w = 1; w < 8; ++w) v = fmaxf(v, smax[w * 16 + threadIdx.x]);
        rowmaxG[b * Tc + Mbase + threadIdx.x] = v;   // rows exclusive to this block
    }
}

// ---------------------------------------------------------------------------
// 3) g = softmax_T( tanh(maxpre) ) for each of 2B rows (rowmax||colmax -> gq||ga)
// ---------------------------------------------------------------------------
__global__ void __launch_bounds__(256)
k_softmax(const float* __restrict__ src, float* __restrict__ dst) {
    __shared__ float red[256];
    const int tid = threadIdx.x;
    const float* x = src + (size_t)blockIdx.x * Tc;
    float m = -__builtin_inff();
    for (int i = tid; i < Tc; i += 256) m = fmaxf(m, tanhf(x[i]));
    red[tid] = m; __syncthreads();
    for (int s = 128; s > 0; s >>= 1) {
        if (tid < s) red[tid] = fmaxf(red[tid], red[tid + s]);
        __syncthreads();
    }
    m = red[0]; __syncthreads();
    float sum = 0.0f;
    for (int i = tid; i < Tc; i += 256) sum += __expf(tanhf(x[i]) - m);
    red[tid] = sum; __syncthreads();
    for (int s = 128; s > 0; s >>= 1) {
        if (tid < s) red[tid] += red[tid + s];
        __syncthreads();
    }
    const float inv = 1.0f / red[0];
    float* y = dst + (size_t)blockIdx.x * Tc;
    for (int i = tid; i < Tc; i += 256) y[i] = __expf(tanhf(x[i]) - m) * inv;
}

// ---------------------------------------------------------------------------
// 4) q_out[b,:] = sum_t q[b,t,:]*gq[b,t] ; a_out likewise. grid (B,2), block D
// ---------------------------------------------------------------------------
__global__ void __launch_bounds__(256)
k_pool(const float* __restrict__ q, const float* __restrict__ a,
       const float* __restrict__ gq, float* __restrict__ out) {
    const int b = blockIdx.x, which = blockIdx.y, d = threadIdx.x;
    const float* src = which ? a : q;
    const float* w   = gq + (size_t)which * Bc * Tc + (size_t)b * Tc;
    float acc = 0.0f;
    for (int t = 0; t < Tc; ++t)
        acc += src[(size_t)(b * Tc + t) * Dc + d] * w[t];
    out[(size_t)which * Bc * Dc + (size_t)b * Dc + d] = acc;
}

// ---------------------------------------------------------------------------
extern "C" void kernel_launch(void* const* d_in, const int* in_sizes, int n_in,
                              void* d_out, int out_size, void* d_ws, size_t ws_size,
                              hipStream_t stream) {
    const float* q = (const float*)d_in[0];   // [B,T,D] f32
    const float* a = (const float*)d_in[1];   // [B,T,D] f32
    const float* U = (const float*)d_in[2];   // [D,D]   f32
    float* out = (float*)d_out;               // [2,B,D] f32

    // workspace layout (~34.2 MB)
    _Float16* Uh      = (_Float16*)d_ws;                      // D*D f16, col-major
    _Float16* qUh     = Uh + (size_t)Dc * Dc;                 // B*T*D f16
    _Float16* ah      = qUh + (size_t)Bc * Tc * Dc;           // B*T*D f16
    float*    rowmaxG = (float*)(ah + (size_t)Bc * Tc * Dc);  // B*T
    float*    colmaxG = rowmaxG + (size_t)Bc * Tc;            // B*T (follows rowmax)
    float*    gq      = colmaxG + (size_t)Bc * Tc;            // 2*B*T (gq then ga)

    k_prepU<<<(Dc * Dc + 255) / 256, 256, 0, stream>>>(U, Uh);
    k_convA<<<(Bc * Tc * Dc / 8 + 255) / 256, 256, 0, stream>>>(a, ah);
    k_initNegInf<<<(Bc * Tc + 255) / 256, 256, 0, stream>>>(colmaxG, Bc * Tc);
    k_qU<<<dim3(Tc / 16, Dc / 64, Bc), 128, 0, stream>>>(q, Uh, qUh);
    k_G<<<dim3(Tc / 16, Bc), 256, 0, stream>>>(ah, qUh, rowmaxG, colmaxG);
    k_softmax<<<2 * Bc, 256, 0, stream>>>(rowmaxG, gq);   // rowmax||colmax -> gq||ga
    k_pool<<<dim3(Bc, 2), 256, 0, stream>>>(q, a, gq, out);
}
